// NER_67869073211815
// MI455X (gfx1250) — matrix operand order repaired and checked by reading, not compiled, not run on hardware
//
#include <hip/hip_runtime.h>
#include <hip/hip_bf16.h>

typedef __attribute__((ext_vector_type(16))) __bf16 v16bf;
typedef __attribute__((ext_vector_type(8)))  __bf16 v8bf;
typedef __attribute__((ext_vector_type(4)))  __bf16 v4bf;
typedef __attribute__((ext_vector_type(8)))  float  v8f;
typedef __attribute__((ext_vector_type(4)))  float  v4f;
typedef __attribute__((ext_vector_type(2)))  float  v2f;
typedef __attribute__((ext_vector_type(4)))  unsigned int u32x4;
typedef __attribute__((ext_vector_type(8)))  int i32x8;
typedef __attribute__((ext_vector_type(4)))  int i32x4;

#define S_LEN 512
#define BATCH 32
#define HID   128
#define XH    768
#define NTAG  33

__device__ __forceinline__ float sigm(float x) { return 1.f / (1.f + __expf(-x)); }

// generic pointer -> LDS byte offset (AS(3) pointers are 32-bit LDS offsets)
__device__ __forceinline__ unsigned lds_off_u32(void* p) {
  return (unsigned)(unsigned long long)(__attribute__((address_space(3))) void*)p;
}

// Issue a TDM 2D tile load: 32 rows x 512 f32, row stride S_LEN*512 elements,
// from xg + t*512 into LDS at lds_byte_off. D# packing per CDNA5 ISA ch.8.
__device__ __forceinline__ void tdm_issue_xg(const float* xg, int t,
                                             unsigned lds_byte_off) {
  unsigned long long ga = (unsigned long long)(const void*)(xg + (size_t)t * 512);
  u32x4 g0;
  g0[0] = 1u;                                            // count=1 (valid D#)
  g0[1] = lds_byte_off;                                  // lds_addr
  g0[2] = (unsigned)(ga & 0xFFFFFFFFull);                // global_addr[31:0]
  g0[3] = (unsigned)((ga >> 32) & 0x1FFFFFFull) | (2u << 30); // addr[56:32]|type=2
  i32x8 g1;
  g1[0] = (int)(2u << 16);            // workgroup_mask=0, data_size=2 (4B)
  g1[1] = (int)(512u << 16);          // tensor_dim0[15:0]=512 in bits 63:48
  g1[2] = (int)(32u << 16);           // tensor_dim1[15:0]=32 in bits 95:80
  g1[3] = (int)(512u << 16);          // tile_dim0=512 in bits 127:112
  g1[4] = 32;                         // tile_dim1=32, tile_dim2=0
  g1[5] = (int)(512u * 512u);         // tensor_dim0_stride = 262144 elements
  g1[6] = 0;                          // stride hi / dim1_stride lo
  g1[7] = 0;
  i32x4 z4 = (i32x4)0;
#if __clang_major__ >= 23
  i32x8 z8 = (i32x8)0;
  __builtin_amdgcn_tensor_load_to_lds(g0, g1, z4, z4, z8, 0);
#else
  __builtin_amdgcn_tensor_load_to_lds(g0, g1, z4, z4, 0);
#endif
}

// ---------------------------------------------------------------------------
// Y[m,n] = sum_k X[m*K+k] * W[n*K+k] + bias[n]   (bf16 WMMA, f32 accumulate)
// Block = 8 waves sharing one 16-row A strip staged in LDS (bf16, padded pitch
// so fragment reads are LDS-bank-conflict-free). One wave per 16x16 C tile.
// ---------------------------------------------------------------------------
__global__ __launch_bounds__(256) void
gemm_xwt_bf16(const float* __restrict__ X, const float* __restrict__ W,
              const float* __restrict__ bias, float* __restrict__ Y,
              int K, int N, int ldC, int nTiles) {
  __shared__ __bf16 sA[16 * (XH + 8)];      // max K=768, pitch K+8 (~24.3 KB)
  int tid = threadIdx.x, lane = tid & 31, wv = tid >> 5;
  int lane15 = lane & 15, hi = lane >> 4;
  int m0 = blockIdx.x * 16;
  int lda = K + 8;
  // cooperative A staging: f32 global (b128) -> bf16 LDS (b64 packed stores)
  for (int r = 0; r < 16; ++r) {
    const float* xr = X + (size_t)(m0 + r) * K;
    for (int k4 = tid * 4; k4 < K; k4 += 1024) {
      v4f x = *(const v4f*)(xr + k4);
      v4bf p;
      p[0] = (__bf16)x[0]; p[1] = (__bf16)x[1];
      p[2] = (__bf16)x[2]; p[3] = (__bf16)x[3];
      *(v4bf*)(sA + r * lda + k4) = p;
    }
  }
  __syncthreads();
  int nTile = blockIdx.y * 8 + wv;
  if (nTile < nTiles) {
    int n = nTile * 16 + lane15;
    int nc = (n < N) ? n : 0;               // clamp OOB reads on padded N tiles
    const float* wrow = W + (size_t)nc * K;
    const __bf16* arow = sA + lane15 * lda;
    v8f acc = (v8f)0.f;
    for (int k0 = 0; k0 < K; k0 += 32) {
      // A fragment: elem e -> K = k0 + hi*16 + e
      v8bf a0 = *(const v8bf*)(arow + k0 + hi * 16);
      v8bf a1 = *(const v8bf*)(arow + k0 + hi * 16 + 8);
      v16bf a;
#pragma unroll
      for (int j = 0; j < 8; ++j) { a[j] = a0[j]; a[8 + j] = a1[j]; }
      // B fragment: elem e -> K = k0 + hi*16 + e  (4x b128 + cvt)
      const float* wp = wrow + k0 + hi * 16;
      __builtin_prefetch(wp + 32, 0, 3);    // next k-chunk -> global_prefetch
      v4f w0 = *(const v4f*)(wp + 0);
      v4f w1 = *(const v4f*)(wp + 4);
      v4f w2 = *(const v4f*)(wp + 8);
      v4f w3 = *(const v4f*)(wp + 12);
      v16bf b;
#pragma unroll
      for (int e = 0; e < 4; ++e) {
        b[e]      = (__bf16)w0[e];
        b[4 + e]  = (__bf16)w1[e];
        b[8 + e]  = (__bf16)w2[e];
        b[12 + e] = (__bf16)w3[e];
      }
      acc = __builtin_amdgcn_wmma_f32_16x16x32_bf16(false, a, false, b, (short)0, acc,
                                                    false, false);
    }
    if (n < N) {
      float bvv = bias[n];
      int mb = m0 + hi * 8;
#pragma unroll
      for (int r = 0; r < 8; ++r) Y[(size_t)(mb + r) * ldC + n] = acc[r] + bvv;
    }
  }
}

// ---------------------------------------------------------------------------
// Bidirectional LSTM scan. grid=2 (dir), block=1024 (32 waves).
// h,c,gates resident in LDS; per-step 32x512 gate GEMM via
// V_WMMA_F32_16X16X4_F32 (full fp32 recurrence). The per-step xg slice
// (32 rows x 512 f32, 1MB row stride) is double-buffered into LDS by the
// Tensor Data Mover: wave 0 waits TENSORcnt for the current buffer, then
// issues the next step's TDM load, overlapping DMA with the WMMA chain.
// ---------------------------------------------------------------------------
__global__ __launch_bounds__(1024) void
lstm_scan(const float* __restrict__ xg_f, const float* __restrict__ xg_b,
          const float* __restrict__ Whh_f, const float* __restrict__ Whh_b,
          const float* __restrict__ h0, const float* __restrict__ c0,
          float* __restrict__ hf, float* __restrict__ hb) {
  __shared__ float sh_h[BATCH * HID];       // 16 KB
  __shared__ float sh_c[BATCH * HID];       // 16 KB
  __shared__ float sh_g[BATCH * 4 * HID];   // 64 KB
  __shared__ float sh_xg[2][BATCH * 512];   // 128 KB (TDM double buffer)
  int dir = blockIdx.x;
  const float* xg  = dir ? xg_b  : xg_f;
  const float* Whh = dir ? Whh_b : Whh_f;
  float* hout      = dir ? hb    : hf;
  int tid = threadIdx.x, lane = tid & 31, wv = tid >> 5;
  int lane15 = lane & 15, hi = lane >> 4;
  unsigned xgoff0 = lds_off_u32(&sh_xg[0][0]);
  unsigned xgoff1 = lds_off_u32(&sh_xg[1][0]);
  for (int e = tid; e < BATCH * HID; e += 1024) {
    sh_h[e] = h0[dir * BATCH * HID + e];
    sh_c[e] = c0[dir * BATCH * HID + e];
  }
  if (wv == 0) {                            // prime buffer 0 with step 0's slice
    int t0 = dir ? (S_LEN - 1) : 0;
    tdm_issue_xg(xg, t0, xgoff0);
  }
  __syncthreads();
  int ncol = wv * 16 + lane15;              // gate column 0..511
  const float* wrow = Whh + (size_t)ncol * HID;
  for (int step = 0; step < S_LEN; ++step) {
    int t = dir ? (S_LEN - 1 - step) : step;
    int buf = step & 1;
    if (wv == 0) {
      __builtin_amdgcn_s_wait_tensorcnt(0); // TDM for this step has landed
      if (step + 1 < S_LEN) {               // prefetch next step into other buf
        int tn = dir ? (S_LEN - 2 - step) : (step + 1);
        tdm_issue_xg(xg, tn, buf ? xgoff0 : xgoff1);
      }
    }
    __syncthreads();                        // publish TDM data to all waves
    const float* xgs = sh_xg[buf];
    v8f acc0 = (v8f)0.f, acc1 = (v8f)0.f;   // two independent M-tile chains
    for (int ki = 0; ki < 32; ++ki) {
      v2f b = *(const v2f*)(wrow + ki * 4 + hi * 2);          // shared B frag
      v2f a0 = *(const v2f*)(sh_h + lane15 * HID + ki * 4 + hi * 2);
      v2f a1 = *(const v2f*)(sh_h + (16 + lane15) * HID + ki * 4 + hi * 2);
      acc0 = __builtin_amdgcn_wmma_f32_16x16x4_f32(false, a0, false, b, (short)0, acc0,
                                                   false, false);
      acc1 = __builtin_amdgcn_wmma_f32_16x16x4_f32(false, a1, false, b, (short)0, acc1,
                                                   false, false);
    }
#pragma unroll
    for (int r = 0; r < 8; ++r) {
      int mm0 = hi * 8 + r;                 // batch rows of tile 0 / tile 1
      int mm1 = 16 + mm0;
      sh_g[mm0 * 512 + ncol] = acc0[r] + xgs[mm0 * 512 + ncol];
      sh_g[mm1 * 512 + ncol] = acc1[r] + xgs[mm1 * 512 + ncol];
    }
    __syncthreads();
    for (int e = tid; e < BATCH * HID; e += 1024) {
      int bb = e >> 7, hh = e & 127;
      float ig = sigm(sh_g[bb * 512 + hh]);
      float fg = sigm(sh_g[bb * 512 + 128 + hh]);
      float gg = tanhf(sh_g[bb * 512 + 256 + hh]);
      float og = sigm(sh_g[bb * 512 + 384 + hh]);
      float cn = fg * sh_c[e] + ig * gg;
      float hn = og * tanhf(cn);
      sh_c[e] = cn; sh_h[e] = hn;
      hout[((size_t)(bb * S_LEN + t)) * HID + hh] = hn;
    }
    __syncthreads();
  }
}

__global__ __launch_bounds__(256) void
add_vec4(const float* __restrict__ a, const float* __restrict__ b,
         float* __restrict__ y, int n4) {
  int i = blockIdx.x * blockDim.x + threadIdx.x;
  if (i < n4) {
    v4f av = *(const v4f*)(a + i * 4);
    v4f bv = *(const v4f*)(b + i * 4);
    *(v4f*)(y + i * 4) = av + bv;
  }
}

// ---------------------------------------------------------------------------
// Flash-style attention, dk=16. grid=(S/128, heads, batch), block=256 (8 waves).
// Q fragment is invariant across key tiles -> hoisted to registers.
// QK^T and PV via f32 WMMA; online softmax state in LDS per wave.
// ---------------------------------------------------------------------------
__global__ __launch_bounds__(256) void
attn_fa(const float* __restrict__ Q, const float* __restrict__ Kt,
        const float* __restrict__ V, float* __restrict__ O) {
  __shared__ float sq[128 * 16];            // 8 KB (pre-scaled q rows)
  __shared__ float sk[16 * 16];
  __shared__ float sv[16 * 16];
  __shared__ float sp[8 * 16 * 16];         // per-wave score/prob tile
  __shared__ float smx[8 * 16], sl[8 * 16], sal[8 * 16];
  int tid = threadIdx.x, lane = tid & 31, wv = tid >> 5;
  int lane15 = lane & 15, hi = lane >> 4;
  int b = blockIdx.z, h = blockIdx.y, q0 = blockIdx.x * 128;
  const float scale = 0.25f;                // 1/sqrt(16)
  for (int e = tid; e < 128 * 16; e += 256) {
    int r = e >> 4, d = e & 15;
    sq[e] = Q[((size_t)(b * S_LEN + q0 + r)) * HID + h * 16 + d] * scale;
  }
  if (tid < 128) { smx[tid] = -1e30f; sl[tid] = 0.f; }
  __syncthreads();
  v2f qf[4];                                // hoisted Q fragment (16 k-values)
#pragma unroll
  for (int ki = 0; ki < 4; ++ki)
    qf[ki] = *(const v2f*)(sq + (wv * 16 + lane15) * 16 + ki * 4 + hi * 2);
  v8f acc = (v8f)0.f;
  for (int kt = 0; kt < 32; ++kt) {
    {
      int r = tid >> 4, d = tid & 15;       // 256 threads -> one 16x16 tile each
      size_t gi = ((size_t)(b * S_LEN + kt * 16 + r)) * HID + h * 16 + d;
      sk[tid] = Kt[gi];
      sv[tid] = V[gi];
    }
    __syncthreads();
    // S = q @ k^T
    v8f s = (v8f)0.f;
#pragma unroll
    for (int ki = 0; ki < 4; ++ki) {
      v2f bb = *(const v2f*)(sk + lane15 * 16 + ki * 4 + hi * 2);
      s = __builtin_amdgcn_wmma_f32_16x16x4_f32(false, qf[ki], false, bb, (short)0, s,
                                                false, false);
    }
    float* pw = sp + wv * 256;
#pragma unroll
    for (int r = 0; r < 8; ++r) pw[(hi * 8 + r) * 16 + lane15] = s[r];
    // online softmax, one row per lane (lanes 0..15)
    if (lane < 16) {
      float* pr = pw + lane * 16;
      float mx = pr[0];
      for (int j = 1; j < 16; ++j) mx = fmaxf(mx, pr[j]);
      float om = smx[wv * 16 + lane];
      float nm = fmaxf(om, mx);
      float sum = 0.f;
      for (int j = 0; j < 16; ++j) { float p = __expf(pr[j] - nm); pr[j] = p; sum += p; }
      float al = __expf(om - nm);
      sal[wv * 16 + lane] = al;
      smx[wv * 16 + lane] = nm;
      sl[wv * 16 + lane] = sl[wv * 16 + lane] * al + sum;
    }
#pragma unroll
    for (int r = 0; r < 8; ++r) acc[r] *= sal[wv * 16 + hi * 8 + r];
    // acc += P @ V
#pragma unroll
    for (int ki = 0; ki < 4; ++ki) {
      v2f a = *(const v2f*)(pw + lane15 * 16 + ki * 4 + hi * 2);
      v2f bb;
      bb[0] = sv[(ki * 4 + hi * 2) * 16 + lane15];
      bb[1] = sv[(ki * 4 + hi * 2 + 1) * 16 + lane15];
      acc = __builtin_amdgcn_wmma_f32_16x16x4_f32(false, a, false, bb, (short)0, acc,
                                                  false, false);
    }
    __syncthreads();
  }
#pragma unroll
  for (int r = 0; r < 8; ++r) {
    int mr = hi * 8 + r;
    float val = acc[r] / sl[wv * 16 + mr];
    O[((size_t)(b * S_LEN + q0 + wv * 16 + mr)) * HID + h * 16 + lane15] = val;
  }
}

// ---------------------------------------------------------------------------
// CRF NLL: one block (64 thr) per sequence. Transition matrix + double-
// buffered alpha in LDS, serial forward over 512 steps.
// ---------------------------------------------------------------------------
__global__ __launch_bounds__(64) void
crf_nll(const float* __restrict__ logits, const int* __restrict__ labels,
        const int* __restrict__ mask, const float* __restrict__ startv,
        const float* __restrict__ endv, const float* __restrict__ trans,
        float* __restrict__ llh) {
  __shared__ float str[NTAG * NTAG];
  __shared__ float alA[NTAG], alB[NTAG];
  __shared__ float red[64];
  __shared__ int   redi[64];
  int b = blockIdx.x, tid = threadIdx.x;
  for (int e = tid; e < NTAG * NTAG; e += 64) str[e] = trans[e];
  const float* em = logits + (size_t)b * S_LEN * NTAG;
  const int* tg = labels + b * S_LEN;
  const int* mk = mask + b * S_LEN;
  float np = 0.f; int cnt = 0;
  for (int t = tid; t < S_LEN; t += 64) {
    int mt = (mk[t] == 1);
    cnt += mt;
    if (t >= 1 && mt) np += trans[tg[t - 1] * NTAG + tg[t]] + em[t * NTAG + tg[t]];
  }
  red[tid] = np; redi[tid] = cnt;
  __syncthreads();
  for (int s = 32; s > 0; s >>= 1) {
    if (tid < s) { red[tid] += red[tid + s]; redi[tid] += redi[tid + s]; }
    __syncthreads();
  }
  if (tid < NTAG) alA[tid] = startv[tid] + em[tid];
  if (tid == 0) {
    int seqlen = redi[0];
    red[0] = red[0] + startv[tg[0]] + em[tg[0]] + endv[tg[seqlen - 1]];
  }
  __syncthreads();
  float* cur = alA; float* nxt = alB;
  for (int t = 1; t < S_LEN; ++t) {
    if (tid < NTAG) {
      float mx = -1e30f;
      for (int i = 0; i < NTAG; ++i) mx = fmaxf(mx, cur[i] + str[i * NTAG + tid]);
      float s = 0.f;
      for (int i = 0; i < NTAG; ++i) s += __expf(cur[i] + str[i * NTAG + tid] - mx);
      float v = mx + __logf(s) + em[t * NTAG + tid];
      nxt[tid] = (mk[t] == 1) ? v : cur[tid];
    }
    __syncthreads();
    float* tmp = cur; cur = nxt; nxt = tmp;
  }
  if (tid == 0) {
    float mx = -1e30f;
    for (int j = 0; j < NTAG; ++j) mx = fmaxf(mx, cur[j] + endv[j]);
    float s = 0.f;
    for (int j = 0; j < NTAG; ++j) s += __expf(cur[j] + endv[j] - mx);
    llh[b] = red[0] - (mx + __logf(s));
  }
}

__global__ __launch_bounds__(32) void
final_reduce(const float* __restrict__ llh, float* __restrict__ out) {
  float s = llh[threadIdx.x];               // 32 threads = 1 wave
  for (int o = 16; o > 0; o >>= 1) s += __shfl_down(s, o, 32);
  if (threadIdx.x == 0) out[0] = -s / 32.f;
}

// ---------------------------------------------------------------------------
extern "C" void kernel_launch(void* const* d_in, const int* in_sizes, int n_in,
                              void* d_out, int out_size, void* d_ws, size_t ws_size,
                              hipStream_t stream) {
  const float* hs    = (const float*)d_in[0];
  const int*   amask = (const int*)d_in[1];
  const int*   labels= (const int*)d_in[2];
  const float* Wih_f = (const float*)d_in[3];
  const float* Whh_f = (const float*)d_in[4];
  const float* b_f   = (const float*)d_in[5];
  const float* Wih_b = (const float*)d_in[6];
  const float* Whh_b = (const float*)d_in[7];
  const float* b_b   = (const float*)d_in[8];
  const float* h0    = (const float*)d_in[9];
  const float* c0    = (const float*)d_in[10];
  const float* Wq    = (const float*)d_in[11];
  const float* bq    = (const float*)d_in[12];
  const float* Wk    = (const float*)d_in[13];
  const float* bk    = (const float*)d_in[14];
  const float* Wv    = (const float*)d_in[15];
  const float* bv    = (const float*)d_in[16];
  const float* Wo    = (const float*)d_in[17];
  const float* bo    = (const float*)d_in[18];
  const float* Wtag  = (const float*)d_in[19];
  const float* btag  = (const float*)d_in[20];
  const float* cst   = (const float*)d_in[21];
  const float* cen   = (const float*)d_in[22];
  const float* ctr   = (const float*)d_in[23];

  float* ws = (float*)d_ws;
  const size_t M = (size_t)BATCH * S_LEN;   // 16384
  // phase-1 layout
  float* xg_f = ws;                         // 8M floats
  float* xg_b = xg_f + M * 512;             // 8M floats
  float* hf   = xg_b + M * 512;             // 2M
  float* hb   = hf + M * HID;               // 2M
  float* xsum = hb + M * HID;               // 2M
  // phase-2 layout (reuses xg region, free after the scan)
  float* q    = ws;
  float* kk   = q  + M * HID;
  float* vv   = kk + M * HID;
  float* oo   = vv + M * HID;
  float* attn = oo + M * HID;
  float* logit= attn + M * HID;
  float* llh  = logit + M * NTAG;

  // 1) input-gate projections (dominant GEMMs, bf16 WMMA, LDS-shared A)
  gemm_xwt_bf16<<<dim3(M / 16, 4), 256, 0, stream>>>(hs, Wih_f, b_f, xg_f, XH, 512, 512, 32);
  gemm_xwt_bf16<<<dim3(M / 16, 4), 256, 0, stream>>>(hs, Wih_b, b_b, xg_b, XH, 512, 512, 32);
  // 2) resident-state bidirectional LSTM scan (fp32 WMMA + TDM prefetch)
  lstm_scan<<<2, 1024, 0, stream>>>(xg_f, xg_b, Whh_f, Whh_b, h0, c0, hf, hb);
  // 3) x = h_f + h_b
  int n4 = (int)(M * HID / 4);
  add_vec4<<<(n4 + 255) / 256, 256, 0, stream>>>(hf, hb, xsum, n4);
  // 4) q/k/v projections
  gemm_xwt_bf16<<<dim3(M / 16, 1), 256, 0, stream>>>(xsum, Wq, bq, q,  HID, HID, HID, 8);
  gemm_xwt_bf16<<<dim3(M / 16, 1), 256, 0, stream>>>(xsum, Wk, bk, kk, HID, HID, HID, 8);
  gemm_xwt_bf16<<<dim3(M / 16, 1), 256, 0, stream>>>(xsum, Wv, bv, vv, HID, HID, HID, 8);
  // 5) flash attention (fp32 WMMA)
  attn_fa<<<dim3(4, 8, 32), 256, 0, stream>>>(q, kk, vv, oo);
  // 6) output projection
  gemm_xwt_bf16<<<dim3(M / 16, 1), 256, 0, stream>>>(oo, Wo, bo, attn, HID, HID, HID, 8);
  // 7) tag logits (N=33 padded to 3 tiles)
  gemm_xwt_bf16<<<dim3(M / 16, 1), 256, 0, stream>>>(attn, Wtag, btag, logit, HID, NTAG, NTAG, 3);
  // 8) CRF forward + numerator per sequence
  crf_nll<<<32, 64, 0, stream>>>(logit, labels, amask, cst, cen, ctr, llh);
  // 9) -mean(llh)
  final_reduce<<<1, 32, 0, stream>>>(llh, (float*)d_out);
  (void)in_sizes; (void)n_in; (void)out_size; (void)ws_size;
}
